// ManualLSTM_18708877541817
// MI455X (gfx1250) — matrix operand order repaired and checked by reading, not compiled, not run on hardware
//
#include <hip/hip_runtime.h>
#include <hip/hip_bf16.h>
#include <math.h>

#define V_    50000
#define E_    256
#define H_    512
#define C_    2
#define B_    64
#define L_    512
#define INP_  768            // E + H
#define NGATE 4
#define NCOL  (NGATE*H_)     // 2048
#define NT    (NCOL/16)      // 128 N-tiles
#define KT_TOT (INP_/32)     // 24 K-tiles total
#define KT_X  (E_/32)        // 8 K-tiles of input part
#define KT_H  (H_/32)        // 16 K-tiles of hidden part
#define MT    (B_/16)        // 4 M-tiles

typedef __attribute__((ext_vector_type(16))) __bf16 v16bf;
typedef __attribute__((ext_vector_type(8)))  float  v8f;

union FragU {
    v16bf v;
    uint4 q[2];
    unsigned short u[16];
};

static __device__ __forceinline__ unsigned short f2bf(float f) {
    union { float f; unsigned int u; } x; x.f = f;
    unsigned int u = x.u;
    if ((u & 0x7fffffffu) > 0x7f800000u) return (unsigned short)((u >> 16) | 0x40u); // NaN
    unsigned int r = 0x7fffu + ((u >> 16) & 1u);                                    // RNE
    return (unsigned short)((u + r) >> 16);
}

// A-matrix (16x32 bf16) fragment k for (lane, half-index i)
static __device__ __forceinline__ int kmapA(int lane, int i) {
    return (i & 7) + ((i >= 8) ? 16 : 0) + (lane >> 4) * 8;
}

// ---------------------------------------------------------------------------
// Pack the 4 gate weight matrices (H x INP, row-major, fp32) into the
// B-operand (32x16, bf16) fragment layout: idx = ((nt*KT_TOT + kt)*32 + lane)*16 + i
// B[k][n] = W_gate[j][k], n = gate*H + j.
// ---------------------------------------------------------------------------
__global__ void pack_w_kernel(const float* __restrict__ Wf, const float* __restrict__ Wi,
                              const float* __restrict__ Wc, const float* __restrict__ Wo,
                              unsigned short* __restrict__ Wpk) {
    int idx = blockIdx.x * blockDim.x + threadIdx.x;
    if (idx >= NT * KT_TOT * 512) return;
    int i    = idx & 15;
    int lane = (idx >> 4) & 31;
    int rest = idx >> 9;                 // nt*KT_TOT + kt
    int kt   = rest % KT_TOT;
    int nt   = rest / KT_TOT;
    int n    = nt * 16 + (lane & 15);
    int k    = kt * 32 + kmapA(lane, i);
    int gate = n >> 9;
    int j    = n & (H_ - 1);
    const float* W = (gate == 0) ? Wf : (gate == 1) ? Wi : (gate == 2) ? Wc : Wo;
    Wpk[idx] = f2bf(W[j * INP_ + k]);
}

// ---------------------------------------------------------------------------
// Embedding gather + bf16 convert directly into A-operand (16x32) fragment
// layout per timestep: idx = (((t*MT + mt)*KT_X + kt)*32 + lane)*16 + i
// A[m][k] = embed[x[b][t]][k], m = b.
// ---------------------------------------------------------------------------
__global__ void pack_x_kernel(const int* __restrict__ x, const float* __restrict__ embed,
                              unsigned short* __restrict__ Apk) {
    int idx = blockIdx.x * blockDim.x + threadIdx.x;
    if (idx >= L_ * MT * KT_X * 512) return;
    int i    = idx & 15;
    int lane = (idx >> 4) & 31;
    int rest = idx >> 9;                 // (t*MT + mt)*KT_X + kt
    int kt   = rest % KT_X;
    int mt   = (rest / KT_X) % MT;
    int t    = rest / (KT_X * MT);
    int b    = mt * 16 + (lane & 15);
    int e    = kt * 32 + kmapA(lane, i);
    int tok  = x[b * L_ + t];
    Apk[idx] = f2bf(embed[(size_t)tok * E_ + e]);
}

__global__ void init_state_kernel(float* h, float* c, float* hsum, unsigned short* hpk) {
    int idx = blockIdx.x * blockDim.x + threadIdx.x;
    if (idx < B_ * H_) { h[idx] = 0.f; c[idx] = 0.f; hsum[idx] = 0.f; hpk[idx] = 0; }
}

// ---------------------------------------------------------------------------
// One LSTM timestep. 32 workgroups x 256 threads (8 wave32).
// Workgroup `wg` owns hidden columns [wg*16, wg*16+16) for all 4 gates:
// wave w -> gate = w&3, M-tile pair = (w>>2)*2. Each wave does 2 C-tiles of
// 16x16, K = 768 (24 bf16 WMMA k-steps each), then gates meet in LDS for the
// elementwise cell update. h is re-packed to bf16 A-fragments (double buffer).
// ---------------------------------------------------------------------------
__global__ __launch_bounds__(256)
void lstm_step_kernel(const unsigned short* __restrict__ Wpk,
                      const unsigned short* __restrict__ Apk,
                      const unsigned short* __restrict__ hpk_in,
                      unsigned short* __restrict__ hpk_out,
                      float* __restrict__ h, float* __restrict__ c,
                      float* __restrict__ hsum,
                      const float* __restrict__ bfv, const float* __restrict__ biv,
                      const float* __restrict__ bcv, const float* __restrict__ bov,
                      const unsigned char* __restrict__ mask, int t) {
    __shared__ float gl[NGATE][B_][16];     // 16 KB gate pre-activations

    const int wg   = blockIdx.x;            // 0..31
    const int tid  = threadIdx.x;
    const int wave = tid >> 5;
    const int lane = tid & 31;
    const int gate = wave & 3;
    const int mt0  = (wave >> 2) * 2;       // 0 or 2
    const int nt   = gate * 32 + wg;        // 0..127

    v8f acc0 = {0.f,0.f,0.f,0.f,0.f,0.f,0.f,0.f};
    v8f acc1 = {0.f,0.f,0.f,0.f,0.f,0.f,0.f,0.f};

    const unsigned short* wp = Wpk + ((size_t)nt * KT_TOT) * 512 + (size_t)lane * 16;
    const unsigned short* ax = Apk + ((size_t)t * MT) * KT_X * 512;

    for (int kt = 0; kt < KT_TOT; ++kt) {
        FragU bfr;
        const uint4* bq = (const uint4*)(wp + (size_t)kt * 512);
        bfr.q[0] = bq[0]; bfr.q[1] = bq[1];
        if (kt + 1 < KT_TOT) __builtin_prefetch(wp + (size_t)(kt + 1) * 512, 0, 1);

        const unsigned short *pa0, *pa1;
        if (kt < KT_X) {
            pa0 = ax + (((size_t)(mt0    ) * KT_X + kt) * 32 + lane) * 16;
            pa1 = ax + (((size_t)(mt0 + 1) * KT_X + kt) * 32 + lane) * 16;
        } else {
            int kth = kt - KT_X;
            pa0 = hpk_in + (((size_t)(mt0    ) * KT_H + kth) * 32 + lane) * 16;
            pa1 = hpk_in + (((size_t)(mt0 + 1) * KT_H + kth) * 32 + lane) * 16;
        }
        FragU a0, a1;
        a0.q[0] = ((const uint4*)pa0)[0]; a0.q[1] = ((const uint4*)pa0)[1];
        a1.q[0] = ((const uint4*)pa1)[0]; a1.q[1] = ((const uint4*)pa1)[1];

        acc0 = __builtin_amdgcn_wmma_f32_16x16x32_bf16(false, a0.v, false, bfr.v,
                                                       (short)0, acc0, false, false);
        acc1 = __builtin_amdgcn_wmma_f32_16x16x32_bf16(false, a1.v, false, bfr.v,
                                                       (short)0, acc1, false, false);
    }

    // C/D layout: m = r + (lane>>4)*8, n = lane&15
    const int nl = lane & 15;
    const int mr = (lane >> 4) * 8;
    #pragma unroll
    for (int r = 0; r < 8; ++r) {
        gl[gate][mt0 * 16 + mr + r][nl]       = acc0[r];
        gl[gate][(mt0 + 1) * 16 + mr + r][nl] = acc1[r];
    }
    __syncthreads();

    // Elementwise cell update: 64x16 elements, 4 per thread.
    #pragma unroll
    for (int q2 = 0; q2 < 4; ++q2) {
        int e    = tid + q2 * 256;
        int b    = e >> 4;
        int nloc = e & 15;
        int j    = wg * 16 + nloc;
        float fp = gl[0][b][nloc] + bfv[j];
        float ip = gl[1][b][nloc] + biv[j];
        float cp = gl[2][b][nloc] + bcv[j];
        float op = gl[3][b][nloc] + bov[j];
        float fg = 1.f / (1.f + __expf(-fp));
        float ig = 1.f / (1.f + __expf(-ip));
        float ct = tanhf(cp);
        float og = 1.f / (1.f + __expf(-op));
        int   off  = b * H_ + j;
        float cold = c[off];
        float cn   = fg * cold + ig * ct;      // c updated unconditionally (ref semantics)
        float hn   = og * tanhf(cn);
        float m    = (float)mask[b * L_ + t];
        float ha   = hn * m + h[off] * (1.f - m);
        c[off]     = cn;
        h[off]     = ha;
        hsum[off] += ha * m;
        // repack h into bf16 A-fragment layout for the next step
        int kth   = j >> 5;
        int kk    = j & 31;
        int hi    = (kk >> 3) & 1;
        int ii    = (kk & 7) + ((kk >= 16) ? 8 : 0);
        int lane2 = (b & 15) + hi * 16;
        int mt2   = b >> 4;
        hpk_out[(((size_t)mt2 * KT_H + kth) * 32 + lane2) * 16 + ii] = f2bf(ha);
    }
}

// ---------------------------------------------------------------------------
// Masked mean-pool + tiny FC head. One block per batch row.
// out[0:128] = logits (B x C), out[128:] = rep (B x H).
// ---------------------------------------------------------------------------
__global__ __launch_bounds__(256)
void finalize_kernel(const float* __restrict__ hsum, const unsigned char* __restrict__ mask,
                     const float* __restrict__ Wfc, const float* __restrict__ bfc,
                     float* __restrict__ out) {
    __shared__ float red[256];
    __shared__ float repS[H_];
    int b = blockIdx.x;
    int tid = threadIdx.x;

    float s = 0.f;
    for (int t = tid; t < L_; t += 256) s += (float)mask[b * L_ + t];
    red[tid] = s; __syncthreads();
    for (int st = 128; st > 0; st >>= 1) { if (tid < st) red[tid] += red[tid + st]; __syncthreads(); }
    float denom = (red[0] < 1e-9f) ? 1e-9f : red[0];
    __syncthreads();

    for (int j = tid; j < H_; j += 256) {
        float r = hsum[b * H_ + j] / denom;
        repS[j] = r;
        out[B_ * C_ + b * H_ + j] = r;
    }
    __syncthreads();

    for (int cls = 0; cls < C_; ++cls) {
        float p = 0.f;
        for (int j = tid; j < H_; j += 256) p += repS[j] * Wfc[cls * H_ + j];
        red[tid] = p; __syncthreads();
        for (int st = 128; st > 0; st >>= 1) { if (tid < st) red[tid] += red[tid + st]; __syncthreads(); }
        if (tid == 0) out[b * C_ + cls] = red[0] + bfc[cls];
        __syncthreads();
    }
}

extern "C" void kernel_launch(void* const* d_in, const int* in_sizes, int n_in,
                              void* d_out, int out_size, void* d_ws, size_t ws_size,
                              hipStream_t stream) {
    const int*           x     = (const int*)d_in[0];
    const unsigned char* mask  = (const unsigned char*)d_in[1];
    const float*         embed = (const float*)d_in[2];
    const float*         Wf    = (const float*)d_in[3];
    const float*         bf    = (const float*)d_in[4];
    const float*         Wi    = (const float*)d_in[5];
    const float*         bi    = (const float*)d_in[6];
    const float*         Wc    = (const float*)d_in[7];
    const float*         bc    = (const float*)d_in[8];
    const float*         Wo    = (const float*)d_in[9];
    const float*         bo    = (const float*)d_in[10];
    const float*         Wfc   = (const float*)d_in[11];
    const float*         bfc   = (const float*)d_in[12];
    float*               out   = (float*)d_out;

    char*  ws  = (char*)d_ws;
    size_t off = 0;
    auto carve = [&](size_t bytes) -> void* {
        void* p = ws + off;
        off += (bytes + 255) & ~(size_t)255;
        return p;
    };
    unsigned short* Wpk  = (unsigned short*)carve((size_t)NT * KT_TOT * 512 * 2);      // 3 MB
    unsigned short* Apk  = (unsigned short*)carve((size_t)L_ * MT * KT_X * 512 * 2);   // 16 MB
    unsigned short* hpkA = (unsigned short*)carve((size_t)B_ * H_ * 2);
    unsigned short* hpkB = (unsigned short*)carve((size_t)B_ * H_ * 2);
    float*          h    = (float*)carve((size_t)B_ * H_ * 4);
    float*          c    = (float*)carve((size_t)B_ * H_ * 4);
    float*          hs   = (float*)carve((size_t)B_ * H_ * 4);

    pack_w_kernel<<<(NT * KT_TOT * 512 + 255) / 256, 256, 0, stream>>>(Wf, Wi, Wc, Wo, Wpk);
    pack_x_kernel<<<(L_ * MT * KT_X * 512 + 255) / 256, 256, 0, stream>>>(x, embed, Apk);
    init_state_kernel<<<(B_ * H_ + 255) / 256, 256, 0, stream>>>(h, c, hs, hpkA);

    for (int t = 0; t < L_; ++t) {
        const unsigned short* hin  = (t & 1) ? hpkB : hpkA;
        unsigned short*       hout = (t & 1) ? hpkA : hpkB;
        lstm_step_kernel<<<32, 256, 0, stream>>>(Wpk, Apk, hin, hout, h, c, hs,
                                                 bf, bi, bc, bo, mask, t);
    }
    finalize_kernel<<<B_, 256, 0, stream>>>(hs, mask, Wfc, bfc, out);
}
